// DocumentEncoder_50139448213883
// MI455X (gfx1250) — matrix operand order repaired and checked by reading, not compiled, not run on hardware
//
#include <hip/hip_runtime.h>

typedef __bf16 bf16_t;
typedef __attribute__((ext_vector_type(16))) __bf16 v16bf;
typedef __attribute__((ext_vector_type(8)))  float  v8f;
typedef __attribute__((ext_vector_type(4)))  unsigned int u32x4;

#define B_   32
#define S_   512
#define D_   768
#define H_   384
#define H3_  1152
#define TOK_ (B_*S_)     // 16384 token rows
#define MT_  (TOK_/16)   // 1024 M-tiles
#define KD_  (D_/32)     // 24 k-steps over D
#define KH_  (H_/32)     // 12 k-steps over H
#define NT3_ (H3_/16)    // 72 gate N-tiles
#define NT2H_ (768/16)   // 48 attn N-tiles
#define HP_  392         // padded bf16 row stride for h shadow (bank-conflict avoidance)

static __device__ __forceinline__ float sigmoidf_(float x) {
  return 1.0f / (1.0f + __expf(-x));
}
static __device__ __forceinline__ float fast_tanhf_(float x) {
#if __has_builtin(__builtin_amdgcn_tanh_f32)
  return __builtin_amdgcn_tanh_f32(x);       // native CDNA5 v_tanh_f32 (TRANS op)
#else
  const float e = __expf(2.0f * x);
  return 1.0f - 2.0f / (e + 1.0f);
#endif
}
static __device__ __forceinline__ v8f zero8() {
  v8f z;
#pragma unroll
  for (int i = 0; i < 8; ++i) z[i] = 0.0f;
  return z;
}

// ---------------------------------------------------------------------------
// Repack fp32 row-major matrices into WMMA fragment-linear bf16 buffers.
// mode 0 (A-matrix): frag row m = src row; K pattern = interleaved octets
// mode 1 (B-matrix): frag col n = src row; K pattern = interleaved octets
// dst layout: [tile][kstep][lane][16] bf16  (32 B per lane, fully coalesced)
// ---------------------------------------------------------------------------
__global__ __launch_bounds__(256) void repack_kernel(
    const float* __restrict__ src, bf16_t* __restrict__ dst,
    int tiles, int ksteps, int srcK, int mode)
{
  int gid = blockIdx.x * blockDim.x + threadIdx.x;
  int total = tiles * ksteps * 32;
  if (gid >= total) return;
  int lane = gid & 31;
  int kk   = (gid >> 5) % ksteps;
  int tile = gid / (ksteps * 32);
  int r    = tile * 16 + (lane & 15);
  int half = lane >> 4;
  const float* __restrict__ s = src + (size_t)r * srcK;
  v16bf o;
  if (mode == 0) {
    int k0 = kk * 32 + half * 8;
#pragma unroll
    for (int i = 0; i < 8; ++i) {
      o[i]     = (bf16_t)s[k0 + i];
      o[8 + i] = (bf16_t)s[k0 + 16 + i];
    }
  } else {
    int k0 = kk * 32 + half * 8;
#pragma unroll
    for (int i = 0; i < 8; ++i) {
      o[i]     = (bf16_t)s[k0 + i];
      o[8 + i] = (bf16_t)s[k0 + 16 + i];
    }
  }
  *(v16bf*)(dst + (size_t)gid * 16) = o;
}

// ---------------------------------------------------------------------------
// Input projection GEMM: xw = x @ W_ih^T + b_ih   [16384 x 768] x [768 x 1152]
// grid (1024 M-tiles, 2 directions), 8 waves/block, 9 N-tiles per wave.
// B-fragment loads grouped per k-step so one clause/wait covers the WMMA burst.
// ---------------------------------------------------------------------------
__global__ __launch_bounds__(256) void gemm1_kernel(
    const bf16_t* __restrict__ afrag,
    const bf16_t* __restrict__ bfrag_f, const bf16_t* __restrict__ bfrag_b,
    const float* __restrict__ bih_f, const float* __restrict__ bih_b,
    float* __restrict__ xw_f, float* __restrict__ xw_b)
{
  const int mtile = blockIdx.x;
  const int dir   = blockIdx.y;
  const bf16_t* __restrict__ bfrag = dir ? bfrag_b : bfrag_f;
  const float*  __restrict__ bih   = dir ? bih_b   : bih_f;
  float* __restrict__ xw           = dir ? xw_b    : xw_f;
  const int wave = threadIdx.x >> 5;
  const int lane = threadIdx.x & 31;

  const v16bf* __restrict__ A  = (const v16bf*)afrag + (size_t)mtile * KD_ * 32 + lane;
  const v16bf* __restrict__ Bv = (const v16bf*)bfrag + lane;

  v8f acc[9];
#pragma unroll
  for (int i = 0; i < 9; ++i) acc[i] = zero8();

  for (int kk = 0; kk < KD_; ++kk) {
    v16bf a = A[kk * 32];
    if (kk + 1 < KD_) __builtin_prefetch(&A[(kk + 1) * 32], 0, 3);
    v16bf bb[9];
#pragma unroll
    for (int nt = 0; nt < 9; ++nt)
      bb[nt] = Bv[((size_t)(wave * 9 + nt) * KD_ + kk) * 32];
#pragma unroll
    for (int nt = 0; nt < 9; ++nt)
      acc[nt] = __builtin_amdgcn_wmma_f32_16x16x32_bf16(
          false, a, false, bb[nt], (short)0, acc[nt], false, false);
  }

  const int n16 = lane & 15, half = lane >> 4;
#pragma unroll
  for (int nt = 0; nt < 9; ++nt) {
    const int n = (wave * 9 + nt) * 16 + n16;
    const float bias = bih[n];
#pragma unroll
    for (int j = 0; j < 8; ++j) {
      const int row = mtile * 16 + j + 8 * half;
      xw[(size_t)row * H3_ + n] = acc[nt][j] + bias;
    }
  }
}

// ---------------------------------------------------------------------------
// Persistent GRU recurrence. grid (2 dirs, 2 batch-halves); each block owns 16
// batch rows, iterates 512 timesteps: WMMA h@W_hh^T into LDS, then fused gates.
// ---------------------------------------------------------------------------
__global__ __launch_bounds__(256) void gru_rec_kernel(
    const bf16_t* __restrict__ bfragWhh_f, const bf16_t* __restrict__ bfragWhh_b,
    const float* __restrict__ bhh_f, const float* __restrict__ bhh_b,
    const float* __restrict__ xw_f, const float* __restrict__ xw_b,
    float* __restrict__ out)
{
  const int dir   = blockIdx.x;
  const int bhalf = blockIdx.y;
  const bf16_t* __restrict__ bfrag = dir ? bfragWhh_b : bfragWhh_f;
  const float*  __restrict__ bhh   = dir ? bhh_b : bhh_f;
  const float*  __restrict__ xw    = dir ? xw_b  : xw_f;

  __shared__ float  hS[16 * H_];    // fp32 hidden state
  __shared__ bf16_t hB[16 * HP_];   // bf16 shadow for WMMA A-fragments (padded)
  __shared__ float  gh[16 * H3_];   // gate pre-activations h@W_hh^T + b_hh

  const int tid = threadIdx.x;
  for (int i = tid; i < 16 * H_;  i += 256) hS[i] = 0.0f;
  for (int i = tid; i < 16 * HP_; i += 256) hB[i] = (bf16_t)0.0f;
  __syncthreads();

  const int wave = tid >> 5, lane = tid & 31;
  const int m16 = lane & 15, half = lane >> 4;
  const v16bf* __restrict__ Bv = (const v16bf*)bfrag + lane;

  for (int t = 0; t < S_; ++t) {
    const int tt = dir ? (S_ - 1 - t) : t;

    v8f acc[9];
#pragma unroll
    for (int i = 0; i < 9; ++i) acc[i] = zero8();

    for (int kk = 0; kk < KH_; ++kk) {
      union { v16bf v; u32x4 q[2]; } au;
      const int k0 = kk * 32 + half * 8;
      au.q[0] = *(const u32x4*)&hB[m16 * HP_ + k0];
      au.q[1] = *(const u32x4*)&hB[m16 * HP_ + k0 + 16];
      v16bf bb[9];
#pragma unroll
      for (int nt = 0; nt < 9; ++nt)
        bb[nt] = Bv[((size_t)(wave * 9 + nt) * KH_ + kk) * 32];
#pragma unroll
      for (int nt = 0; nt < 9; ++nt)
        acc[nt] = __builtin_amdgcn_wmma_f32_16x16x32_bf16(
            false, au.v, false, bb[nt], (short)0, acc[nt], false, false);
    }
#pragma unroll
    for (int nt = 0; nt < 9; ++nt) {
      const int n = (wave * 9 + nt) * 16 + m16;
      const float bias = bhh[n];
#pragma unroll
      for (int j = 0; j < 8; ++j)
        gh[(j + 8 * half) * H3_ + n] = acc[nt][j] + bias;
    }
    __syncthreads();

    for (int idx = tid; idx < 16 * H_; idx += 256) {
      const int mm = idx / H_;
      const int d  = idx - mm * H_;
      const int bb = bhalf * 16 + mm;
      const size_t tok = (size_t)bb * S_ + tt;
      const float* __restrict__ xrow = xw + tok * H3_;
      const float r  = sigmoidf_(xrow[d]          + gh[mm * H3_ + d]);
      const float z  = sigmoidf_(xrow[H_ + d]     + gh[mm * H3_ + H_ + d]);
      const float nn = fast_tanhf_(xrow[2 * H_ + d] + r * gh[mm * H3_ + 2 * H_ + d]);
      const float hnew = (1.0f - z) * nn + z * hS[idx];
      hS[idx] = hnew;
      hB[mm * HP_ + d] = (bf16_t)hnew;
      out[tok * 768 + dir * H_ + d] = hnew;
    }
    __syncthreads();
  }
}

// ---------------------------------------------------------------------------
// Fused attention scores: score[row] = sum_n context[n]*tanh((out@W_attn^T)[row,n]+b[n])
// One block per M-tile; 8 waves x 6 N-tiles; shuffle-reduce + LDS float atomics.
// ---------------------------------------------------------------------------
__global__ __launch_bounds__(256) void attn_scores_kernel(
    const bf16_t* __restrict__ afragOut, const bf16_t* __restrict__ bfragWattn,
    const float* __restrict__ battn, const float* __restrict__ ctx,
    float* __restrict__ scores)
{
  const int mtile = blockIdx.x;
  __shared__ float ssc[16];
  if (threadIdx.x < 16) ssc[threadIdx.x] = 0.0f;
  __syncthreads();

  const int wave = threadIdx.x >> 5, lane = threadIdx.x & 31;
  const int n16 = lane & 15, half = lane >> 4;
  const v16bf* __restrict__ A  = (const v16bf*)afragOut + (size_t)mtile * KD_ * 32 + lane;
  const v16bf* __restrict__ Bv = (const v16bf*)bfragWattn + lane;

  v8f acc[6];
#pragma unroll
  for (int i = 0; i < 6; ++i) acc[i] = zero8();

  for (int kk = 0; kk < KD_; ++kk) {
    v16bf a = A[kk * 32];
    if (kk + 1 < KD_) __builtin_prefetch(&A[(kk + 1) * 32], 0, 3);
    v16bf bb[6];
#pragma unroll
    for (int nt = 0; nt < 6; ++nt)
      bb[nt] = Bv[((size_t)(wave * 6 + nt) * KD_ + kk) * 32];
#pragma unroll
    for (int nt = 0; nt < 6; ++nt)
      acc[nt] = __builtin_amdgcn_wmma_f32_16x16x32_bf16(
          false, a, false, bb[nt], (short)0, acc[nt], false, false);
  }

#pragma unroll
  for (int nt = 0; nt < 6; ++nt) {
    const int n = (wave * 6 + nt) * 16 + n16;
    const float bias = battn[n];
    const float c    = ctx[n];
#pragma unroll
    for (int j = 0; j < 8; ++j) {
      float u = fast_tanhf_(acc[nt][j] + bias) * c;
      u += __shfl_xor(u, 1, 32);
      u += __shfl_xor(u, 2, 32);
      u += __shfl_xor(u, 4, 32);
      u += __shfl_xor(u, 8, 32);
      if (n16 == 0) atomicAdd(&ssc[j + 8 * half], u);
    }
  }
  __syncthreads();
  if (threadIdx.x < 16) scores[mtile * 16 + threadIdx.x] = ssc[threadIdx.x];
}

// ---------------------------------------------------------------------------
// Softmax over S + weighted pooling: doc[b,:] = sum_s softmax(score)[s]*out[b,s,:]
// ---------------------------------------------------------------------------
__global__ __launch_bounds__(256) void attn_pool_kernel(
    const float* __restrict__ scores, const float* __restrict__ out,
    float* __restrict__ doc)
{
  const int b = blockIdx.x;
  const int tid = threadIdx.x;
  __shared__ float p[S_];
  __shared__ float red[256];

  float m = -3.4e38f;
  for (int i = tid; i < S_; i += 256) m = fmaxf(m, scores[b * S_ + i]);
  red[tid] = m;
  __syncthreads();
  for (int s = 128; s > 0; s >>= 1) {
    if (tid < s) red[tid] = fmaxf(red[tid], red[tid + s]);
    __syncthreads();
  }
  const float mx = red[0];
  __syncthreads();

  float sum = 0.0f;
  for (int i = tid; i < S_; i += 256) {
    const float e = __expf(scores[b * S_ + i] - mx);
    p[i] = e;
    sum += e;
  }
  red[tid] = sum;
  __syncthreads();
  for (int s = 128; s > 0; s >>= 1) {
    if (tid < s) red[tid] += red[tid + s];
    __syncthreads();
  }
  const float inv = 1.0f / red[0];
  __syncthreads();

  for (int k = tid; k < 768; k += 256) {
    float a = 0.0f;
    const float* __restrict__ ob = out + (size_t)b * S_ * 768 + k;
    for (int s = 0; s < S_; ++s) a += p[s] * ob[(size_t)s * 768];
    doc[b * 768 + k] = a * inv;
  }
}

// ---------------------------------------------------------------------------
extern "C" void kernel_launch(void* const* d_in, const int* in_sizes, int n_in,
                              void* d_out, int out_size, void* d_ws, size_t ws_size,
                              hipStream_t stream)
{
  (void)in_sizes; (void)n_in; (void)out_size; (void)ws_size;
  const float* ip    = (const float*)d_in[0];
  const float* Wih_f = (const float*)d_in[1];
  const float* Whh_f = (const float*)d_in[2];
  const float* bih_f = (const float*)d_in[3];
  const float* bhh_f = (const float*)d_in[4];
  const float* Wih_b = (const float*)d_in[5];
  const float* Whh_b = (const float*)d_in[6];
  const float* bih_b = (const float*)d_in[7];
  const float* bhh_b = (const float*)d_in[8];
  const float* Wattn = (const float*)d_in[9];
  const float* battn = (const float*)d_in[10];
  const float* ctx   = (const float*)d_in[11];
  float* doc = (float*)d_out;

  char* ws = (char*)d_ws;
  size_t off = 0;
  auto alloc = [&](size_t bytes) -> char* {
    char* p = ws + off;
    off += (bytes + 255) & ~(size_t)255;
    return p;
  };
  bf16_t* afrag    = (bf16_t*)alloc((size_t)TOK_ * D_ * 2);   // ip A-frags, reused for out A-frags
  bf16_t* bfWih_f  = (bf16_t*)alloc((size_t)H3_ * D_ * 2);
  bf16_t* bfWih_b  = (bf16_t*)alloc((size_t)H3_ * D_ * 2);
  bf16_t* bfWhh_f  = (bf16_t*)alloc((size_t)H3_ * H_ * 2);
  bf16_t* bfWhh_b  = (bf16_t*)alloc((size_t)H3_ * H_ * 2);
  bf16_t* bfWattn  = (bf16_t*)alloc((size_t)768 * 768 * 2);
  float*  xw_f     = (float*)alloc((size_t)TOK_ * H3_ * 4);
  float*  xw_b     = (float*)alloc((size_t)TOK_ * H3_ * 4);
  float*  outbuf   = (float*)alloc((size_t)TOK_ * 768 * 4);
  float*  scores   = (float*)alloc((size_t)TOK_ * 4);

  auto blocks = [](int total) { return (total + 255) / 256; };

  // 1) repack activations + weights into WMMA fragment buffers
  repack_kernel<<<blocks(MT_ * KD_ * 32), 256, 0, stream>>>(ip, afrag, MT_, KD_, D_, 0);
  repack_kernel<<<blocks(NT3_ * KD_ * 32), 256, 0, stream>>>(Wih_f, bfWih_f, NT3_, KD_, D_, 1);
  repack_kernel<<<blocks(NT3_ * KD_ * 32), 256, 0, stream>>>(Wih_b, bfWih_b, NT3_, KD_, D_, 1);
  repack_kernel<<<blocks(NT3_ * KH_ * 32), 256, 0, stream>>>(Whh_f, bfWhh_f, NT3_, KH_, H_, 1);
  repack_kernel<<<blocks(NT3_ * KH_ * 32), 256, 0, stream>>>(Whh_b, bfWhh_b, NT3_, KH_, H_, 1);
  repack_kernel<<<blocks(NT2H_ * KD_ * 32), 256, 0, stream>>>(Wattn, bfWattn, NT2H_, KD_, 768, 1);

  // 2) input projections for both directions (WMMA GEMM)
  gemm1_kernel<<<dim3(MT_, 2), 256, 0, stream>>>(afrag, bfWih_f, bfWih_b,
                                                 bih_f, bih_b, xw_f, xw_b);

  // 3) persistent GRU recurrence (2 dirs x 2 batch halves)
  gru_rec_kernel<<<dim3(2, 2), 256, 0, stream>>>(bfWhh_f, bfWhh_b, bhh_f, bhh_b,
                                                 xw_f, xw_b, outbuf);

  // 4) repack BiGRU output into A-fragments (reuse afrag buffer)
  repack_kernel<<<blocks(MT_ * KD_ * 32), 256, 0, stream>>>(outbuf, afrag, MT_, KD_, 768, 0);

  // 5) fused attention score GEMM + tanh + context dot
  attn_scores_kernel<<<MT_, 256, 0, stream>>>(afrag, bfWattn, battn, ctx, scores);

  // 6) softmax + weighted pooling
  attn_pool_kernel<<<B_, 256, 0, stream>>>(scores, outbuf, doc);
}
